// CapsuleNet_4329327034711
// MI455X (gfx1250) — compile-verified, matrix operand order checked by reading, not run on hardware
//
#include <hip/hip_runtime.h>

// ---------------------------------------------------------------------------
// CapsuleNet forward for MI455X (gfx1250, wave32, WMMA + TDM).
// Heavy convs lowered to implicit GEMM on v_wmma_f32_16x16x32_bf16 with
// 128x128 block tiles (8 waves x 32x64 -> 8 WMMA per wave per k-step) and
// double-buffered LDS.  All dense 2D operand streams (conv weights, decoder
// activations/weights) are DMA'd into LDS by the Tensor Data Mover
// (tensor_load_to_lds, fenced with s_wait_tensorcnt); only the scattered
// im2col gathers stay on the VALU/VMEM path.  Dynamic routing is fused into
// one workgroup per (class,batch) using the 320KB/WGP LDS.
// ---------------------------------------------------------------------------

typedef __bf16 bf16_t;
typedef __attribute__((ext_vector_type(16))) __bf16 v16bf;
typedef __attribute__((ext_vector_type(8)))  float  v8f;
typedef __attribute__((ext_vector_type(4)))  unsigned int uint4v;
typedef __attribute__((ext_vector_type(8)))  int    int8v;
typedef __attribute__((ext_vector_type(4)))  int    int4v;

union FragAB { v16bf v; uint4 q[2]; };

// Build a 16x32 bf16 WMMA operand fragment from an LDS tile stored
// row-major (row = M for A, row = N for B; 32 K-contiguous bf16 per row).
// Per ISA 16-bit layout: lane L (row = L&15) holds K = 8*(L>>4)..+7 in
// VGPRs 0-3 and K = 16+8*(L>>4)..+7 in VGPRs 4-7  ->  two ds_load_b128.
__device__ __forceinline__ v16bf load_frag(const bf16_t* s, int row, int lane) {
    FragAB f;
    const int hq = (lane >> 4);                       // 0 or 1
    const uint4* p = (const uint4*)(s + row * 32);    // 4 uint4 per row
    f.q[0] = p[hq];
    f.q[1] = p[2 + hq];
    return f.v;
}

__device__ __forceinline__ v8f zero_v8f() {
    v8f z = {0.f, 0.f, 0.f, 0.f, 0.f, 0.f, 0.f, 0.f};
    return z;
}

#define WMMA_BF16(A, B, C) \
    __builtin_amdgcn_wmma_f32_16x16x32_bf16(false, (A), false, (B), (short)0, (C), false, false)

#define MMA_STEP(As, Bs)                                                      \
    do {                                                                      \
        v16bf a0 = load_frag((As), wm * 32 +      (lane & 15), lane);         \
        v16bf a1 = load_frag((As), wm * 32 + 16 + (lane & 15), lane);         \
        v16bf b0 = load_frag((Bs), wn * 64 +      (lane & 15), lane);         \
        v16bf b1 = load_frag((Bs), wn * 64 + 16 + (lane & 15), lane);         \
        v16bf b2 = load_frag((Bs), wn * 64 + 32 + (lane & 15), lane);         \
        v16bf b3 = load_frag((Bs), wn * 64 + 48 + (lane & 15), lane);         \
        acc[0][0] = WMMA_BF16(a0, b0, acc[0][0]);                             \
        acc[0][1] = WMMA_BF16(a0, b1, acc[0][1]);                             \
        acc[0][2] = WMMA_BF16(a0, b2, acc[0][2]);                             \
        acc[0][3] = WMMA_BF16(a0, b3, acc[0][3]);                             \
        acc[1][0] = WMMA_BF16(a1, b0, acc[1][0]);                             \
        acc[1][1] = WMMA_BF16(a1, b1, acc[1][1]);                             \
        acc[1][2] = WMMA_BF16(a1, b2, acc[1][2]);                             \
        acc[1][3] = WMMA_BF16(a1, b3, acc[1][3]);                             \
    } while (0)

// ---------------------------------------------------------------------------
// TDM: DMA a 2D bf16 tile (tileK x tileN rows, row pitch strideK elements)
// from global memory into LDS.  D# per CDNA5 ISA §8: group0 = {flags, lds
// addr, 57b global addr, type=2}; group1 = {data_size=2B, tensor dims
// (remaining extent from tile start; OOB reads return zero), tile dims,
// dim0 stride}.  Issued by one wave (EXEC ignored by TDM); completion
// fenced with s_wait_tensorcnt 0.
// ---------------------------------------------------------------------------
__device__ __forceinline__ void tdm_load_tile_b16(
    const bf16_t* gaddr, unsigned lds_off,
    unsigned tensorK, unsigned tensorN, unsigned strideK,
    unsigned tileK, unsigned tileN) {
    unsigned long long ga = (unsigned long long)(size_t)gaddr;
    uint4v g0;
    g0[0] = 1u;                                          // count=1 (valid), user D#
    g0[1] = lds_off;                                     // lds_addr (bytes)
    g0[2] = (unsigned)(ga & 0xffffffffu);                // global_addr[31:0]
    g0[3] = (unsigned)((ga >> 32) & 0x01ffffffu) | (2u << 30);  // addr[56:32], type=2
    int8v g1;
    g1[0] = (int)(1u << 16);                             // data_size=1 -> 2 bytes
    g1[1] = (int)((tensorK & 0xffffu) << 16);            // tensor_dim0[15:0]
    g1[2] = (int)(((tensorK >> 16) & 0xffffu) | ((tensorN & 0xffffu) << 16));
    g1[3] = (int)(((tensorN >> 16) & 0xffffu) | (tileK << 16));   // tile_dim0
    g1[4] = (int)tileN;                                  // tile_dim1 (tile_dim2=0)
    g1[5] = (int)strideK;                                // tensor_dim0_stride[31:0]
    g1[6] = 0;                                           // stride hi, dim1_stride lo
    g1[7] = 0;
    int4v z4 = {0, 0, 0, 0};
#if __clang_major__ >= 23
    int8v z8 = {0, 0, 0, 0, 0, 0, 0, 0};
    __builtin_amdgcn_tensor_load_to_lds(g0, g1, z4, z4, z8, 0);
#else
    __builtin_amdgcn_tensor_load_to_lds(g0, g1, z4, z4, 0);
#endif
}

// ---------------------------------------------------------------------------
// Weight packing: fp32 -> bf16, dest layout [N][Kpad] (K contiguous),
// zero padding for k >= Kvalid.  trans=1 means source is stored (K, N).
// ---------------------------------------------------------------------------
__global__ void pack_w_kernel(const float* __restrict__ src, bf16_t* __restrict__ dst,
                              int N, int Kp, int Kv, int trans) {
    int i = blockIdx.x * 256 + threadIdx.x;
    if (i >= N * Kp) return;
    int n = i / Kp, k = i % Kp;
    float v = 0.f;
    if (k < Kv) v = trans ? src[k * N + n] : src[n * Kv + k];
    dst[i] = (bf16_t)v;
}

// ---------------------------------------------------------------------------
// conv1 implicit GEMM:  M = B*400 (b,oh,ow), K = 243->256 (c,r,s), N = 256.
// A: fp32 im2col gather (VALU path); B: TDM tile DMA.
// ---------------------------------------------------------------------------
__global__ __launch_bounds__(256) void conv1_gemm_kernel(
    const float* __restrict__ x, const bf16_t* __restrict__ Wp,
    const float* __restrict__ bias, bf16_t* __restrict__ hbf) {
    __shared__ bf16_t A_s[2][128 * 32];
    __shared__ bf16_t B_s[2][128 * 32];
    const int tid  = threadIdx.x;
    const int lane = tid & 31;
    const int wv   = tid >> 5;
    const int wm   = wv >> 1, wn = wv & 1;       // 4x2 wave grid, 32x64 tiles
    const int m0   = blockIdx.x * 128;
    const int n0   = blockIdx.y * 128;
    const int kk   = tid & 31;                   // fixed k-column per thread (A)
    const int rr0  = tid >> 5;                   // base row 0..7 (A)

    // Hoisted row decode: addr = baseA[j] + koff(k), invariant over K loop.
    int baseA[16];
    #pragma unroll
    for (int j = 0; j < 16; ++j) {
        int m = m0 + rr0 + 8 * j;
        int b = m / 400, t = m - b * 400;
        int oh = t / 20, ow = t - oh * 20;
        baseA[j] = b * 2352 + oh * 28 + ow;      // b*3*28*28 + oh*28 + ow
    }

    v8f acc[2][4];
    #pragma unroll
    for (int i = 0; i < 2; ++i)
        #pragma unroll
        for (int j = 0; j < 4; ++j) acc[i][j] = zero_v8f();

    auto loadA = [&](int k0, int buf) {
        int k = k0 + kk;
        bool valid = (k < 243);
        int koff = 0;
        if (valid) {                              // one div chain per step
            int c = k / 81, t2 = k - c * 81;
            int r = t2 / 9, s = t2 - r * 9;
            koff = c * 784 + r * 28 + s;          // c*28*28 + r*28 + s
        }
        #pragma unroll
        for (int j = 0; j < 16; ++j) {
            float v = valid ? x[baseA[j] + koff] : 0.f;
            A_s[buf][(rr0 + 8 * j) * 32 + kk] = (bf16_t)v;
        }
    };
    auto tdmB = [&](int k0, int buf) {
        if (tid < 32)
            tdm_load_tile_b16(Wp + (size_t)n0 * 256 + k0,
                              (unsigned)(size_t)&B_s[buf][0],
                              256u - k0, 128u, 256u, 32u, 128u);
    };

    loadA(0, 0); tdmB(0, 0);
    if (tid < 32) __builtin_amdgcn_s_wait_tensorcnt(0);
    __syncthreads();
    const int NK = 8;
    for (int kc = 0; kc < NK; ++kc) {
        int cur = kc & 1;
        if (kc + 1 < NK) {
            loadA((kc + 1) * 32, cur ^ 1);
            tdmB((kc + 1) * 32, cur ^ 1);        // DMA overlaps the WMMAs below
        }
        MMA_STEP(A_s[cur], B_s[cur]);
        if (tid < 32) __builtin_amdgcn_s_wait_tensorcnt(0);
        __syncthreads();
    }

    // Epilogue: D layout -> lane = N, VGPR r -> M = 8*(lane>=16)+r
    float bv[4];
    #pragma unroll
    for (int fn = 0; fn < 4; ++fn) bv[fn] = bias[n0 + wn * 64 + fn * 16 + (lane & 15)];
    #pragma unroll
    for (int fm = 0; fm < 2; ++fm)
    #pragma unroll
    for (int r = 0; r < 8; ++r) {
        int m = m0 + wm * 32 + fm * 16 + ((lane >> 4) << 3) + r;
        int b = m / 400, t = m - b * 400;        // t = oh*20+ow
        size_t obase = (size_t)b * 102400 + t;   // b*256*400 + t
        #pragma unroll
        for (int fn = 0; fn < 4; ++fn) {
            int n = n0 + wn * 64 + fn * 16 + (lane & 15);
            float v = fmaxf(acc[fm][fn][r] + bv[fn], 0.f);
            hbf[obase + (size_t)n * 400] = (bf16_t)v;
        }
    }
}

// ---------------------------------------------------------------------------
// Primary-caps implicit GEMM: M = B*36, K = 256*81 = 20736, N = 256.
// A: bf16 conv1 output (NCHW 20x20), stride-2 im2col gather.
// B: weight tiles (128 rows x 32 k, pitch 20736) DMA'd by the TDM.
// Epilogue scatters fp32 into routing layout p[b][r=cap*36+oh*6+ow][i=n&7].
// ---------------------------------------------------------------------------
__global__ __launch_bounds__(256) void prim_gemm_kernel(
    const bf16_t* __restrict__ hbf, const bf16_t* __restrict__ Wp,
    const float* __restrict__ bias, float* __restrict__ p_ws) {
    __shared__ bf16_t A_s[2][128 * 32];
    __shared__ bf16_t B_s[2][128 * 32];
    const int tid  = threadIdx.x;
    const int lane = tid & 31;
    const int wv   = tid >> 5;
    const int wm   = wv >> 1, wn = wv & 1;
    const int m0   = blockIdx.x * 128;
    const int n0   = blockIdx.y * 128;
    const int kk   = tid & 31;
    const int rr0  = tid >> 5;

    int baseA[16];
    #pragma unroll
    for (int j = 0; j < 16; ++j) {
        int m = m0 + rr0 + 8 * j;
        int b = m / 36, t = m - b * 36;
        int oh = t / 6, ow = t - oh * 6;
        baseA[j] = b * 102400 + (2 * oh) * 20 + 2 * ow;   // b*256*400 + ...
    }

    v8f acc[2][4];
    #pragma unroll
    for (int i = 0; i < 2; ++i)
        #pragma unroll
        for (int j = 0; j < 4; ++j) acc[i][j] = zero_v8f();

    auto loadA = [&](int k0, int buf) {
        int k = k0 + kk;
        int c2 = k / 81, t2 = k - c2 * 81;
        int r = t2 / 9, s = t2 - r * 9;
        int koff = c2 * 400 + r * 20 + s;
        #pragma unroll
        for (int j = 0; j < 16; ++j)
            A_s[buf][(rr0 + 8 * j) * 32 + kk] = hbf[baseA[j] + koff];
    };
    // one TDM issue per workgroup (wave 0); tile lands in B_s[buf] as
    // 128 rows x 32 k-contiguous bf16 -- exactly the fragment layout.
    auto tdmB = [&](int k0, int buf) {
        if (tid < 32)
            tdm_load_tile_b16(Wp + (size_t)n0 * 20736 + k0,
                              (unsigned)(size_t)&B_s[buf][0],
                              20736u - k0, 128u, 20736u, 32u, 128u);
    };

    loadA(0, 0); tdmB(0, 0);
    if (tid < 32) __builtin_amdgcn_s_wait_tensorcnt(0);
    __syncthreads();
    const int NK = 20736 / 32;                   // 648 k-steps
    for (int kc = 0; kc < NK; ++kc) {
        int cur = kc & 1;
        if (kc + 1 < NK) {
            loadA((kc + 1) * 32, cur ^ 1);
            tdmB((kc + 1) * 32, cur ^ 1);        // DMA overlaps the WMMAs below
        }
        MMA_STEP(A_s[cur], B_s[cur]);
        if (tid < 32) __builtin_amdgcn_s_wait_tensorcnt(0);
        __syncthreads();
    }

    float bv[4];
    int cap[4], ii[4];
    #pragma unroll
    for (int fn = 0; fn < 4; ++fn) {
        int n = n0 + wn * 64 + fn * 16 + (lane & 15);
        bv[fn] = bias[n]; cap[fn] = n >> 3; ii[fn] = n & 7;
    }
    #pragma unroll
    for (int fm = 0; fm < 2; ++fm)
    #pragma unroll
    for (int r = 0; r < 8; ++r) {
        int m = m0 + wm * 32 + fm * 16 + ((lane >> 4) << 3) + r;
        int b = m / 36, t = m - b * 36;
        size_t rbase = (size_t)b * 1152 + t;     // + cap*36 later
        #pragma unroll
        for (int fn = 0; fn < 4; ++fn)
            p_ws[(rbase + cap[fn] * 36) * 8 + ii[fn]] = acc[fm][fn][r] + bv[fn];
    }
}

// ---------------------------------------------------------------------------
// Fused dynamic routing: one workgroup per (class c, batch b).
// LDS: p (9216 f32), priors (18432 f32), logits/probs (1152 each), scratch.
// ~118 KB dynamic LDS -> fits MI455X's 320 KB WGP LDS.
// ---------------------------------------------------------------------------
__global__ __launch_bounds__(256) void routing_kernel(
    const float* __restrict__ p, const float* __restrict__ route_w,
    float* __restrict__ caps) {
    extern __shared__ float sm[];
    float* p_s  = sm;                   // 1152*8
    float* pri  = p_s  + 1152 * 8;      // 1152*16
    float* lg   = pri  + 1152 * 16;     // 1152
    float* prob = lg   + 1152;          // 1152
    float* red  = prob + 1152;          // 256
    float* sv   = red  + 256;           // 16
    float* vv   = sv   + 16;            // 16

    const int tid = threadIdx.x;
    const int b   = blockIdx.x & 1023;
    const int c   = blockIdx.x >> 10;

    for (int i = tid; i < 1152 * 8; i += 256) p_s[i] = p[(size_t)b * 9216 + i];
    __syncthreads();

    // priors[r][o] = sum_i p[b,r,i] * route_w[c,r,i,o]   (route_w L2-resident)
    for (int r = tid; r < 1152; r += 256) {
        float pv[8];
        #pragma unroll
        for (int i = 0; i < 8; ++i) pv[i] = p_s[r * 8 + i];
        const float* W = route_w + (((size_t)c * 1152 + r) * 8) * 16;
        #pragma unroll
        for (int o = 0; o < 16; ++o) {
            float s = 0.f;
            #pragma unroll
            for (int i = 0; i < 8; ++i) s += pv[i] * W[i * 16 + o];
            pri[r * 16 + o] = s;
        }
    }
    for (int r = tid; r < 1152; r += 256) lg[r] = 0.f;
    __syncthreads();

    for (int it = 0; it < 3; ++it) {
        float lm = -3.0e38f;
        for (int r = tid; r < 1152; r += 256) lm = fmaxf(lm, lg[r]);
        red[tid] = lm; __syncthreads();
        for (int s = 128; s > 0; s >>= 1) {
            if (tid < s) red[tid] = fmaxf(red[tid], red[tid + s]);
            __syncthreads();
        }
        float mx = red[0]; __syncthreads();
        float ls = 0.f;
        for (int r = tid; r < 1152; r += 256) { float e = __expf(lg[r] - mx); prob[r] = e; ls += e; }
        red[tid] = ls; __syncthreads();
        for (int s = 128; s > 0; s >>= 1) {
            if (tid < s) red[tid] += red[tid + s];
            __syncthreads();
        }
        float inv = 1.f / red[0]; __syncthreads();
        int o = tid & 15, rl = tid >> 4;
        float part = 0.f;
        for (int r = rl; r < 1152; r += 16) part += prob[r] * pri[r * 16 + o];
        red[tid] = part * inv; __syncthreads();
        if (tid < 16) {
            float s = 0.f;
            #pragma unroll
            for (int j = 0; j < 16; ++j) s += red[j * 16 + tid];
            sv[tid] = s;
        }
        __syncthreads();
        float sq = 0.f;
        #pragma unroll
        for (int j = 0; j < 16; ++j) sq += sv[j] * sv[j];
        float scale = (sq / (1.f + sq)) * rsqrtf(sq);
        if (tid < 16) vv[tid] = scale * sv[tid];
        __syncthreads();
        if (it < 2) {
            for (int r = tid; r < 1152; r += 256) {
                float d = 0.f;
                #pragma unroll
                for (int oo = 0; oo < 16; ++oo) d += pri[r * 16 + oo] * vv[oo];
                lg[r] += d;
            }
            __syncthreads();
        }
    }
    if (tid < 16) caps[((size_t)b * 2 + c) * 16 + tid] = vv[tid];
}

// ---------------------------------------------------------------------------
// classes (softmax of capsule norms) -> d_out[0..2047]; argmax mask ->
// bf16 decoder input X0 (B x 32).
// ---------------------------------------------------------------------------
__global__ void mask_kernel(const float* __restrict__ caps,
                            float* __restrict__ out_classes,
                            bf16_t* __restrict__ X0) {
    int b = blockIdx.x * 256 + threadIdx.x;
    if (b >= 1024) return;
    float c0[16], c1[16], n0 = 0.f, n1 = 0.f;
    #pragma unroll
    for (int o = 0; o < 16; ++o) { c0[o] = caps[(b * 2 + 0) * 16 + o]; n0 += c0[o] * c0[o]; }
    #pragma unroll
    for (int o = 0; o < 16; ++o) { c1[o] = caps[(b * 2 + 1) * 16 + o]; n1 += c1[o] * c1[o]; }
    n0 = sqrtf(n0); n1 = sqrtf(n1);
    float mx = fmaxf(n0, n1);
    float e0 = __expf(n0 - mx), e1 = __expf(n1 - mx), s = e0 + e1;
    out_classes[b * 2 + 0] = e0 / s;
    out_classes[b * 2 + 1] = e1 / s;
    int cls = (n1 > n0) ? 1 : 0;   // argmax (ties -> class 0, like jnp.argmax)
    #pragma unroll
    for (int j = 0; j < 16; ++j) X0[b * 32 + j]      = (bf16_t)(cls == 0 ? c0[j] : 0.f);
    #pragma unroll
    for (int j = 0; j < 16; ++j) X0[b * 32 + 16 + j] = (bf16_t)(cls == 1 ? c1[j] : 0.f);
}

// ---------------------------------------------------------------------------
// Generic decoder GEMM (M = 1024): C = A(bf16 [M][K]) * Bp(bf16 [N][K]) + bias.
// Both operand tiles DMA'd by the TDM (OOB rows of ragged N zero-filled by
// the D# tensor-dim clamp).  mode 0: ReLU -> bf16.  mode 1: sigmoid -> fp32.
// ---------------------------------------------------------------------------
__global__ __launch_bounds__(256) void dec_gemm_kernel(
    const bf16_t* __restrict__ A, const bf16_t* __restrict__ Bp,
    const float* __restrict__ bias, int N, int K,
    bf16_t* __restrict__ dst_bf, float* __restrict__ dst_f, int mode) {
    __shared__ bf16_t A_s[2][128 * 32];
    __shared__ bf16_t B_s[2][128 * 32];
    const int tid  = threadIdx.x;
    const int lane = tid & 31;
    const int wv   = tid >> 5;
    const int wm   = wv >> 1, wn = wv & 1;
    const int m0   = blockIdx.x * 128;
    const int n0   = blockIdx.y * 128;

    v8f acc[2][4];
    #pragma unroll
    for (int i = 0; i < 2; ++i)
        #pragma unroll
        for (int j = 0; j < 4; ++j) acc[i][j] = zero_v8f();

    auto tdmLoads = [&](int k0, int buf) {
        if (tid < 32) {
            tdm_load_tile_b16(A + (size_t)m0 * K + k0,
                              (unsigned)(size_t)&A_s[buf][0],
                              (unsigned)(K - k0), (unsigned)(1024 - m0),
                              (unsigned)K, 32u, 128u);
            tdm_load_tile_b16(Bp + (size_t)n0 * K + k0,
                              (unsigned)(size_t)&B_s[buf][0],
                              (unsigned)(K - k0), (unsigned)(N - n0),
                              (unsigned)K, 32u, 128u);
        }
    };

    tdmLoads(0, 0);
    if (tid < 32) __builtin_amdgcn_s_wait_tensorcnt(0);
    __syncthreads();
    const int NK = K / 32;
    for (int kc = 0; kc < NK; ++kc) {
        int cur = kc & 1;
        if (kc + 1 < NK) tdmLoads((kc + 1) * 32, cur ^ 1);   // DMA overlaps WMMA
        MMA_STEP(A_s[cur], B_s[cur]);
        if (tid < 32) __builtin_amdgcn_s_wait_tensorcnt(0);
        __syncthreads();
    }

    #pragma unroll
    for (int fn = 0; fn < 4; ++fn) {
        int n = n0 + wn * 64 + fn * 16 + (lane & 15);
        if (n >= N) continue;
        float bv = bias[n];
        #pragma unroll
        for (int fm = 0; fm < 2; ++fm)
        #pragma unroll
        for (int r = 0; r < 8; ++r) {
            int m = m0 + wm * 32 + fm * 16 + ((lane >> 4) << 3) + r;
            float v = acc[fm][fn][r] + bv;
            if (mode == 0) {
                dst_bf[(size_t)m * N + n] = (bf16_t)fmaxf(v, 0.f);
            } else {
                dst_f[(size_t)m * N + n] = 1.f / (1.f + __expf(-v));
            }
        }
    }
}

// ---------------------------------------------------------------------------
// Launcher
// ---------------------------------------------------------------------------
extern "C" void kernel_launch(void* const* d_in, const int* in_sizes, int n_in,
                              void* d_out, int out_size, void* d_ws, size_t ws_size,
                              hipStream_t stream) {
    (void)in_sizes; (void)n_in; (void)out_size; (void)ws_size;
    const float* x       = (const float*)d_in[0];
    const float* conv1_w = (const float*)d_in[1];
    const float* conv1_b = (const float*)d_in[2];
    const float* prim_w  = (const float*)d_in[3];
    const float* prim_b  = (const float*)d_in[4];
    const float* route_w = (const float*)d_in[5];
    const float* dec_w1  = (const float*)d_in[6];
    const float* dec_b1  = (const float*)d_in[7];
    const float* dec_w2  = (const float*)d_in[8];
    const float* dec_b2  = (const float*)d_in[9];
    const float* dec_w3  = (const float*)d_in[10];
    const float* dec_b3  = (const float*)d_in[11];
    float* outp = (float*)d_out;

    // Workspace carve-out (~264 MB total), 256B aligned chunks.
    char* ws = (char*)d_ws;
    size_t off = 0;
    auto carve = [&](size_t bytes) -> void* {
        void* p = ws + off;
        off += (bytes + 255) & ~(size_t)255;
        return p;
    };
    bf16_t* W1p  = (bf16_t*)carve((size_t)256 * 256 * 2);        // conv1 weights packed
    bf16_t* Wpp  = (bf16_t*)carve((size_t)256 * 20736 * 2);      // prim weights packed
    bf16_t* Wd1  = (bf16_t*)carve((size_t)512 * 32 * 2);
    bf16_t* Wd2  = (bf16_t*)carve((size_t)1024 * 512 * 2);
    bf16_t* Wd3  = (bf16_t*)carve((size_t)784 * 1024 * 2);
    bf16_t* hbf  = (bf16_t*)carve((size_t)1024 * 256 * 400 * 2); // conv1 out, bf16 NCHW
    float*  p_ws = (float*) carve((size_t)1024 * 1152 * 8 * 4);  // primary capsules
    float*  caps = (float*) carve((size_t)1024 * 2 * 16 * 4);
    bf16_t* X0   = (bf16_t*)carve((size_t)1024 * 32 * 2);
    bf16_t* X1   = (bf16_t*)carve((size_t)1024 * 512 * 2);
    bf16_t* X2   = (bf16_t*)carve((size_t)1024 * 1024 * 2);

    auto blocks = [](int total) { return dim3((unsigned)((total + 255) / 256)); };

    // 1) pack weights to bf16 [N][Kpad]
    pack_w_kernel<<<blocks(256 * 256),   256, 0, stream>>>(conv1_w, W1p, 256, 256, 243, 0);
    pack_w_kernel<<<blocks(256 * 20736), 256, 0, stream>>>(prim_w,  Wpp, 256, 20736, 20736, 0);
    pack_w_kernel<<<blocks(512 * 32),    256, 0, stream>>>(dec_w1,  Wd1, 512, 32, 32, 1);
    pack_w_kernel<<<blocks(1024 * 512),  256, 0, stream>>>(dec_w2,  Wd2, 1024, 512, 512, 1);
    pack_w_kernel<<<blocks(784 * 1024),  256, 0, stream>>>(dec_w3,  Wd3, 784, 1024, 1024, 1);

    // 2) conv1 implicit GEMM (M=409600, N=256, K=256pad), TDM B-stream
    conv1_gemm_kernel<<<dim3(3200, 2), 256, 0, stream>>>(x, W1p, conv1_b, hbf);

    // 3) primary caps implicit GEMM (M=36864, N=256, K=20736), TDM B-stream
    prim_gemm_kernel<<<dim3(288, 2), 256, 0, stream>>>(hbf, Wpp, prim_b, p_ws);

    // 4) fused dynamic routing: one WG per (class, batch); ~118KB dynamic LDS
    const size_t route_smem = (size_t)(1152 * 8 + 1152 * 16 + 1152 + 1152 + 256 + 16 + 16) * 4;
    routing_kernel<<<2048, 256, route_smem, stream>>>(p_ws, route_w, caps);

    // 5) classes + argmax mask
    mask_kernel<<<4, 256, 0, stream>>>(caps, outp, X0);

    // 6) decoder MLP (fully TDM-fed GEMMs): 32 -> 512 -> 1024 -> 784
    dec_gemm_kernel<<<dim3(8, 4), 256, 0, stream>>>(X0, Wd1, dec_b1, 512,  32,   X1, nullptr, 0);
    dec_gemm_kernel<<<dim3(8, 8), 256, 0, stream>>>(X1, Wd2, dec_b2, 1024, 512,  X2, nullptr, 0);
    dec_gemm_kernel<<<dim3(8, 7), 256, 0, stream>>>(X2, Wd3, dec_b3, 784,  1024, nullptr, outp + 2048, 1);
}
